// MultiScaleAttention5D_47931835023550
// MI455X (gfx1250) — matrix operand order repaired and checked by reading, not compiled, not run on hardware
//
#include <hip/hip_runtime.h>
#include <cmath>

// ---------- types / helpers ----------
typedef __attribute__((ext_vector_type(16))) __bf16   v16bf;
typedef __attribute__((ext_vector_type(8)))  float    v8f;
typedef __attribute__((ext_vector_type(4)))  float    f32x4;
typedef __attribute__((ext_vector_type(4)))  unsigned u32x4;

union FragU { v16bf v; u32x4 q[2]; };

__device__ __forceinline__ v16bf ld_frag(const __bf16* p0, const __bf16* p1) {
  FragU f;
  f.q[0] = *(const u32x4*)p0;   // elements 0..7
  f.q[1] = *(const u32x4*)p1;   // elements 8..15
  return f.v;
}

__device__ __forceinline__ v8f wmma_bf(v16bf a, v16bf b, v8f c) {
  // v_wmma_f32_16x16x32_bf16  D = A(16x32) * B(32x16) + C
  return __builtin_amdgcn_wmma_f32_16x16x32_bf16(false, a, false, b, (short)0, c,
                                                 false, false);
}

__device__ __forceinline__ v8f vzero8() {
  v8f z = {0.f,0.f,0.f,0.f,0.f,0.f,0.f,0.f};
  return z;
}

__device__ __forceinline__ unsigned pack_bf16(float a, float b) {
  unsigned short ua = __builtin_bit_cast(unsigned short, (__bf16)a);
  unsigned short ub = __builtin_bit_cast(unsigned short, (__bf16)b);
  return (unsigned)ua | ((unsigned)ub << 16);
}

// CDNA5 async global->LDS copy (ASYNCcnt path); falls back to VGPR copy.
// Builtin signature (from hipcc diagnostic): first param is
// 'int __attribute__((vector_size(16))) __device__ *'  i.e. v4i in AS(1);
// second param is the LDS destination in AS(3); then imm offset, imm cpol.
#if defined(__gfx1250__) && __has_builtin(__builtin_amdgcn_global_load_async_to_lds_b128)
#define HAVE_ASYNC_LDS 1
typedef int v4i __attribute__((vector_size(16)));
#define AS1 __attribute__((address_space(1)))
#define AS3 __attribute__((address_space(3)))
__device__ __forceinline__ void async_cp16(void* lds, const void* g) {
  __builtin_amdgcn_global_load_async_to_lds_b128((AS1 v4i*)g, (AS3 v4i*)lds,
                                                 0, 0);
}
__device__ __forceinline__ void async_wait0() {
  asm volatile("s_wait_asynccnt 0" ::: "memory");
}
#else
#define HAVE_ASYNC_LDS 0
#endif

// problem constants
// x: [B=2, T=8, C=32, H=64, W=64]; D_model = 256; pixels per batch = 4096
// scale0: 4x4 windows, ch 0..15  -> N0=2048 tokens/batch, d0=256
// scale1: 2x2 windows, ch 16..31 -> N1=8192 tokens/batch, d1=64

// ---------- weight packing into WMMA B-fragment order ----------
// B(32x16 bf16) k order chosen here (A loads use the same): lane l (n=l&15, h=l>>4): elems e=0..15 -> K=h*16+e
// frag array: [cc(8)][ntile(16)][lane(32)][word(8)] u32, word v packs K=h*16+2v, +1
__global__ void pack_w_qkv_kernel(const float* __restrict__ wq,
                                  const float* __restrict__ wk,
                                  const float* __restrict__ wv,
                                  unsigned* __restrict__ fq,
                                  unsigned* __restrict__ fk,
                                  unsigned* __restrict__ fv) {
  int idx = blockIdx.x * 256 + threadIdx.x;          // 8*16*32*8 = 32768 words
  if (idx >= 32768) return;
  int v     = idx & 7;
  int lane  = (idx >> 3) & 31;
  int ntile = (idx >> 8) & 15;
  int cc    = idx >> 12;
  int h = lane >> 4, n = lane & 15;
  int o  = ntile * 16 + n;
  int k0 = cc * 32 + h * 16 + v * 2;
  fq[idx] = pack_bf16(wq[o * 256 + k0], wq[o * 256 + k0 + 1]);
  fk[idx] = pack_bf16(wk[o * 256 + k0], wk[o * 256 + k0 + 1]);
  fv[idx] = pack_bf16(wv[o * 256 + k0], wv[o * 256 + k0 + 1]);
}

// conv weights wo[o][i][ky][kx] -> [tap(9)][cc(8)][ntile(16)][lane(32)][word(8)]
__global__ void pack_w_conv_kernel(const float* __restrict__ wo,
                                   unsigned* __restrict__ fc) {
  int idx = blockIdx.x * 256 + threadIdx.x;          // 9*32768 = 294912 words
  if (idx >= 294912) return;
  int tap = idx >> 15;
  int rem = idx & 32767;
  int v     = rem & 7;
  int lane  = (rem >> 3) & 31;
  int ntile = (rem >> 8) & 15;
  int cc    = rem >> 12;
  int h = lane >> 4, n = lane & 15;
  int o  = ntile * 16 + n;
  int i0 = cc * 32 + h * 16 + v * 2;
  int ky = tap / 3, kx = tap % 3;
  float a = wo[((size_t)(o * 256 + i0)     * 3 + ky) * 3 + kx];
  float b = wo[((size_t)(o * 256 + i0 + 1) * 3 + ky) * 3 + kx];
  fc[idx] = pack_bf16(a, b);
}

// ---------- QKV projection GEMM (16 pixels x 16 outch per wave) ----------
// Out[m=pixel, o] = sum_k x[b][k][pixel] * W[o][k] + bias[o]
// epilogue scatters bf16 directly into token-major attention layouts.
__global__ __launch_bounds__(128, 1)
void qkv_gemm_kernel(const float* __restrict__ x,
                     const unsigned* __restrict__ fq,
                     const unsigned* __restrict__ fk,
                     const unsigned* __restrict__ fv,
                     const float* __restrict__ bq,
                     const float* __restrict__ bk,
                     const float* __restrict__ bv,
                     __bf16* __restrict__ qf0, __bf16* __restrict__ kf0,
                     __bf16* __restrict__ vf0,
                     __bf16* __restrict__ qf1, __bf16* __restrict__ kf1,
                     __bf16* __restrict__ vf1) {
  __shared__ __align__(16) __bf16 xt[16][32];    // A tile: [pixel][channel]
  int tid = threadIdx.x, lane = tid & 31, w = tid >> 5;
  int h = lane >> 4, ln = lane & 15;
  int bt = blockIdx.x;                           // b*256 + pixel-tile
  int b = bt >> 8, ptile = bt & 255;
  int pixbase = ptile * 16;
  int ntile = blockIdx.y * 4 + w;
  const float* xb = x + (size_t)b * 256 * 4096;
  int ch4 = tid >> 2, m4 = (tid & 3) * 4;        // 128 threads cover 32ch x 16pix

  v8f accq = vzero8(), acck = vzero8(), accv = vzero8();
  for (int cc = 0; cc < 8; ++cc) {
    __syncthreads();
    f32x4 v4 = *(const f32x4*)&xb[(size_t)(cc * 32 + ch4) * 4096 + pixbase + m4];
    #pragma unroll
    for (int j = 0; j < 4; ++j) xt[m4 + j][ch4] = (__bf16)v4[j];
    __syncthreads();
    v16bf a = ld_frag(&xt[ln][h * 16], &xt[ln][h * 16 + 8]);
    size_t foff = ((size_t)(cc * 16 + ntile) * 32 + lane) * 8;
    v16bf wqf = ld_frag((const __bf16*)(fq + foff), (const __bf16*)(fq + foff + 4));
    v16bf wkf = ld_frag((const __bf16*)(fk + foff), (const __bf16*)(fk + foff + 4));
    v16bf wvf = ld_frag((const __bf16*)(fv + foff), (const __bf16*)(fv + foff + 4));
    accq = wmma_bf(a, wqf, accq);
    acck = wmma_bf(a, wkf, acck);
    accv = wmma_bf(a, wvf, accv);
  }

  int o = ntile * 16 + ln;
  float biasq = bq[o], biask = bk[o], biasv = bv[o];
  int t = o >> 5, c_ = o & 31;
  #pragma unroll
  for (int r = 0; r < 8; ++r) {
    int p = pixbase + r + 8 * h;
    int hh = p >> 6, ww = p & 63;
    if (c_ < 16) {                               // scale0: 4x4 windows
      int ntok = (t * 16 + (hh >> 2)) * 16 + (ww >> 2);
      int dd   = c_ * 16 + (hh & 3) * 4 + (ww & 3);
      size_t a0 = ((size_t)b * 2048 + ntok) * 256 + dd;
      qf0[a0] = (__bf16)(accq[r] + biasq);
      kf0[a0] = (__bf16)(acck[r] + biask);
      vf0[a0] = (__bf16)(accv[r] + biasv);
    } else {                                     // scale1: 2x2 windows
      int ci = c_ - 16;
      int ntok = (t * 32 + (hh >> 1)) * 32 + (ww >> 1);
      int dd   = ci * 4 + (hh & 1) * 2 + (ww & 1);
      size_t a1 = ((size_t)b * 8192 + ntok) * 64 + dd;
      qf1[a1] = (__bf16)(accq[r] + biasq);
      kf1[a1] = (__bf16)(acck[r] + biask);
      vf1[a1] = (__bf16)(accv[r] + biasv);
    }
  }
}

// ---------- flash attention (per scale) ----------
// 128 threads = 4 waves; each wave owns 16 query rows & all D output cols.
// K tiles of 32 keys; online softmax in f32; scores scaled by sqrt(D) (faithful
// to reference which multiplies by sqrt(d)).
template <int D, int NTOK, int PH, int PW, int OW, int CIBASE>
__global__ __launch_bounds__(128, 1)
void attn_kernel(const __bf16* __restrict__ qf, const __bf16* __restrict__ kf,
                 const __bf16* __restrict__ vf, __bf16* __restrict__ ybf) {
  constexpr int NCH = D / 32;   // K chunks for S = Q K^T
  constexpr int NDC = D / 16;   // output column chunks
  __shared__ __align__(16) __bf16 ktile[32][D];
  __shared__ __align__(16) __bf16 vtileT[D][32];
  __shared__ __align__(16) __bf16 pstage[4][16][32];

  int tid = threadIdx.x, lane = tid & 31, w = tid >> 5;
  int h = lane >> 4, ln = lane & 15;
  int blk = blockIdx.x;
  int b = blk / (NTOK / 64);
  int rowbase = (blk % (NTOK / 64)) * 64;
  const __bf16* qb = qf + (size_t)b * NTOK * D;
  const __bf16* kb = kf + (size_t)b * NTOK * D;
  const __bf16* vb = vf + (size_t)b * NTOK * D;
  const __bf16* qrowp = qb + (size_t)(rowbase + w * 16 + ln) * D;

  v8f o_acc[NDC];
  #pragma unroll
  for (int ch = 0; ch < NDC; ++ch) o_acc[ch] = vzero8();
  float mrow[8], lrow[8];
  #pragma unroll
  for (int r = 0; r < 8; ++r) { mrow[r] = -1e30f; lrow[r] = 0.f; }
  const float sc = sqrtf((float)D);

  for (int k0 = 0; k0 < NTOK; k0 += 32) {
    __syncthreads();
    // K tile -> LDS (straight copy, b128): CDNA5 async global->LDS if available
#if HAVE_ASYNC_LDS
    for (int idx = tid; idx < 32 * D / 8; idx += 128)
      async_cp16((u32x4*)&ktile[0][0] + idx,
                 (const u32x4*)(kb + (size_t)k0 * D) + idx);
#else
    for (int idx = tid; idx < 32 * D / 8; idx += 128)
      ((u32x4*)&ktile[0][0])[idx] = ((const u32x4*)(kb + (size_t)k0 * D))[idx];
#endif
    // V tile -> LDS transposed: vtileT[feature][key]
    for (int idx = tid; idx < 32 * D / 8; idx += 128) {
      int key = idx / (D / 8);
      int f0  = (idx % (D / 8)) * 8;
      u32x4 q4 = ((const u32x4*)(vb + (size_t)k0 * D))[idx];
      const __bf16* e = (const __bf16*)&q4;
      #pragma unroll
      for (int j = 0; j < 8; ++j) vtileT[f0 + j][key] = e[j];
    }
#if HAVE_ASYNC_LDS
    async_wait0();
#endif
    __syncthreads();
    if (k0 + 32 < NTOK)
      __builtin_prefetch(kb + (size_t)(k0 + 32) * D, 0, 1);  // global_prefetch

    // S tiles: keys 0..15 (s0) and 16..31 (s1)
    v8f s0 = vzero8(), s1 = vzero8();
    #pragma unroll
    for (int cc = 0; cc < NCH; ++cc) {
      v16bf a  = ld_frag(qrowp + cc * 32 + h * 16, qrowp + cc * 32 + h * 16 + 8);
      v16bf b0 = ld_frag(&ktile[ln][cc * 32 + h * 16],
                         &ktile[ln][cc * 32 + h * 16 + 8]);
      v16bf b1 = ld_frag(&ktile[16 + ln][cc * 32 + h * 16],
                         &ktile[16 + ln][cc * 32 + h * 16 + 8]);
      s0 = wmma_bf(a, b0, s0);
      s1 = wmma_bf(a, b1, s1);
    }

    // online softmax (row = r + 8*h, values spread across 16-lane half)
    float p0[8], p1[8], corr[8];
    #pragma unroll
    for (int r = 0; r < 8; ++r) {
      float v0 = s0[r] * sc, v1 = s1[r] * sc;
      float mx = fmaxf(v0, v1);
      mx = fmaxf(mx, __shfl_xor(mx, 1, 32));
      mx = fmaxf(mx, __shfl_xor(mx, 2, 32));
      mx = fmaxf(mx, __shfl_xor(mx, 4, 32));
      mx = fmaxf(mx, __shfl_xor(mx, 8, 32));
      float mnew = fmaxf(mrow[r], mx);
      corr[r] = expf(mrow[r] - mnew);
      p0[r] = expf(v0 - mnew);
      p1[r] = expf(v1 - mnew);
      float ps = p0[r] + p1[r];
      ps += __shfl_xor(ps, 1, 32);
      ps += __shfl_xor(ps, 2, 32);
      ps += __shfl_xor(ps, 4, 32);
      ps += __shfl_xor(ps, 8, 32);
      lrow[r] = lrow[r] * corr[r] + ps;
      mrow[r] = mnew;
    }

    // transpose P (C layout -> A layout) through per-wave LDS staging
    #pragma unroll
    for (int r = 0; r < 8; ++r) {
      pstage[w][r + 8 * h][ln]      = (__bf16)p0[r];
      pstage[w][r + 8 * h][16 + ln] = (__bf16)p1[r];
    }
    asm volatile("s_wait_dscnt 0" ::: "memory");
    v16bf pfrag = ld_frag(&pstage[w][ln][h * 16], &pstage[w][ln][h * 16 + 8]);

    // O = diag(corr)*O + P * V
    #pragma unroll
    for (int ch = 0; ch < NDC; ++ch) {
      #pragma unroll
      for (int r = 0; r < 8; ++r) o_acc[ch][r] *= corr[r];
      v16bf vfrag = ld_frag(&vtileT[ch * 16 + ln][h * 16],
                            &vtileT[ch * 16 + ln][h * 16 + 8]);
      o_acc[ch] = wmma_bf(pfrag, vfrag, o_acc[ch]);
    }
  }

  // normalize + scatter back to padded image layout ybf[b][256][66][66]
  float inv[8];
  #pragma unroll
  for (int r = 0; r < 8; ++r) inv[r] = 1.0f / lrow[r];
  #pragma unroll
  for (int ch = 0; ch < NDC; ++ch) {
    int dd = ch * 16 + ln;
    int ci = dd / (PH * PW);
    int pr = dd % (PH * PW);
    int py = pr / PW, px = pr % PW;
    #pragma unroll
    for (int r = 0; r < 8; ++r) {
      int token = rowbase + w * 16 + r + 8 * h;
      int t   = token / (OW * OW);
      int rem = token % (OW * OW);
      int ohh = rem / OW, oww = rem % OW;
      int hh = ohh * PH + py, ww = oww * PW + px;
      int chn = t * 32 + CIBASE + ci;
      ybf[((size_t)b * 256 + chn) * 66 * 66 + (size_t)(hh + 1) * 66 + (ww + 1)] =
          (__bf16)(o_acc[ch][r] * inv[r]);
    }
  }
}

// ---------- 3x3 conv (implicit GEMM) + bias + LeakyReLU ----------
__global__ __launch_bounds__(128, 1)
void conv3x3_kernel(const __bf16* __restrict__ ybf,
                    const unsigned* __restrict__ fc,
                    const float* __restrict__ bo, float* __restrict__ out) {
  __shared__ __align__(16) __bf16 patch[3][18][32];  // [ky][x][ch]
  int tid = threadIdx.x, lane = tid & 31, w = tid >> 5;
  int h = lane >> 4, ln = lane & 15;
  int bt = blockIdx.x;                       // b*256 + tile
  int b = bt >> 8, tile = bt & 255;
  int y = tile >> 2, x0 = (tile & 3) * 16;   // 16 pixels along x, row y
  int ntile = blockIdx.y * 4 + w;

  v8f acc = vzero8();
  for (int cc = 0; cc < 8; ++cc) {
    __syncthreads();
    for (int idx = tid; idx < 1728; idx += 128) {    // 32ch * 3 * 18
      int ch = idx / 54, sp = idx % 54;
      int ry = sp / 18, rx = sp % 18;
      patch[ry][rx][ch] =
          ybf[(((size_t)b * 256 + cc * 32 + ch) * 66 + y + ry) * 66 + x0 + rx];
    }
    __syncthreads();
    #pragma unroll
    for (int tap = 0; tap < 9; ++tap) {
      int ky = tap / 3, kx = tap % 3;
      v16bf a = ld_frag(&patch[ky][ln + kx][h * 16],
                        &patch[ky][ln + kx][h * 16 + 8]);
      size_t foff = (((size_t)(tap * 8 + cc) * 16 + ntile) * 32 + lane) * 8;
      v16bf wf = ld_frag((const __bf16*)(fc + foff),
                         (const __bf16*)(fc + foff + 4));
      acc = wmma_bf(a, wf, acc);
    }
  }

  int o = ntile * 16 + ln;
  float bias = bo[o];
  #pragma unroll
  for (int r = 0; r < 8; ++r) {
    float v = acc[r] + bias;
    v = v >= 0.f ? v : 0.2f * v;             // LeakyReLU(0.2)
    int p = y * 64 + x0 + r + 8 * h;
    out[((size_t)b * 256 + o) * 4096 + p] = v;
  }
}

// ---------- workspace layout (bytes, all 256-aligned) ----------
static const size_t OFF_QF0 = 0;                       // 2*2048*256*2 = 2 MiB
static const size_t OFF_KF0 = 2097152;
static const size_t OFF_VF0 = 4194304;
static const size_t OFF_QF1 = 6291456;                 // 2*8192*64*2 = 2 MiB
static const size_t OFF_KF1 = 8388608;
static const size_t OFF_VF1 = 10485760;
static const size_t OFF_YBF = 12582912;                // 2*256*66*66*2
static const size_t YBF_BYTES = 4460544;
static const size_t OFF_FQ = OFF_YBF + YBF_BYTES;      // 131072 each
static const size_t OFF_FK = OFF_FQ + 131072;
static const size_t OFF_FV = OFF_FK + 131072;
static const size_t OFF_FC = OFF_FV + 131072;          // 1179648

extern "C" void kernel_launch(void* const* d_in, const int* in_sizes, int n_in,
                              void* d_out, int out_size, void* d_ws,
                              size_t ws_size, hipStream_t stream) {
  const float* x  = (const float*)d_in[0];
  const float* wq = (const float*)d_in[1];
  const float* bq = (const float*)d_in[2];
  const float* wk = (const float*)d_in[3];
  const float* bk = (const float*)d_in[4];
  const float* wv = (const float*)d_in[5];
  const float* bv = (const float*)d_in[6];
  const float* wo = (const float*)d_in[7];
  const float* bo = (const float*)d_in[8];
  char* ws = (char*)d_ws;

  __bf16* qf0 = (__bf16*)(ws + OFF_QF0);
  __bf16* kf0 = (__bf16*)(ws + OFF_KF0);
  __bf16* vf0 = (__bf16*)(ws + OFF_VF0);
  __bf16* qf1 = (__bf16*)(ws + OFF_QF1);
  __bf16* kf1 = (__bf16*)(ws + OFF_KF1);
  __bf16* vf1 = (__bf16*)(ws + OFF_VF1);
  __bf16* ybf = (__bf16*)(ws + OFF_YBF);
  unsigned* fq = (unsigned*)(ws + OFF_FQ);
  unsigned* fk = (unsigned*)(ws + OFF_FK);
  unsigned* fv = (unsigned*)(ws + OFF_FV);
  unsigned* fc = (unsigned*)(ws + OFF_FC);

  // zero padded image (borders stay zero for SAME conv)
  (void)hipMemsetAsync(ybf, 0, YBF_BYTES, stream);

  pack_w_qkv_kernel<<<128, 256, 0, stream>>>(wq, wk, wv, fq, fk, fv);
  pack_w_conv_kernel<<<1152, 256, 0, stream>>>(wo, fc);

  qkv_gemm_kernel<<<dim3(512, 4), 128, 0, stream>>>(
      x, fq, fk, fv, bq, bk, bv, qf0, kf0, vf0, qf1, kf1, vf1);

  // scale0: D=256, N=2048/batch, 4x4 windows, OW=16, channels 0..15
  attn_kernel<256, 2048, 4, 4, 16, 0><<<64, 128, 0, stream>>>(qf0, kf0, vf0, ybf);
  // scale1: D=64, N=8192/batch, 2x2 windows, OW=32, channels 16..31
  attn_kernel<64, 8192, 2, 2, 32, 16><<<256, 128, 0, stream>>>(qf1, kf1, vf1, ybf);

  conv3x3_kernel<<<dim3(512, 4), 128, 0, stream>>>(ybf, fc, bo, (float*)d_out);
}
